// PhiLayer_77756087927330
// MI455X (gfx1250) — matrix operand order repaired and checked by reading, not compiled, not run on hardware
//
#include <hip/hip_runtime.h>

typedef __attribute__((ext_vector_type(16))) __bf16 v16bf;
typedef __attribute__((ext_vector_type(8))) float v8f;

#define DIMD   1024
#define NHEADS 16
#define HDIM   64
#define SEQ    2048
#define BATCH  2
#define MTOT   (BATCH * SEQ)
#define GHID   256

#define EPI_GELU 0
#define EPI_GATE 1
#define EPI_BF16 2
#define EPI_F32  3

__device__ __forceinline__ unsigned short f2bf(float f) {
  unsigned int u = __float_as_uint(f);
  u += 0x7FFFu + ((u >> 16) & 1u);           // round-to-nearest-even
  return (unsigned short)(u >> 16);
}
__device__ __forceinline__ unsigned int pack2bf(float a, float b) {
  return ((unsigned int)f2bf(b) << 16) | (unsigned int)f2bf(a);
}

union FragB { v16bf v; unsigned int u[8]; };

// ---- CDNA5 async global->LDS copy (ASYNCcnt-tracked), ISA §15.18 / §08 ----
__device__ __forceinline__ void async_copy_b32(unsigned lds_byte_addr,
                                               const void* gptr) {
  asm volatile("global_load_async_to_lds_b32 %0, %1, off"
               :: "v"(lds_byte_addr),
                  "v"((unsigned long long)(uintptr_t)gptr)
               : "memory");
}
__device__ __forceinline__ void wait_async0() {
  asm volatile("s_wait_asynccnt 0x0" ::: "memory");
}

// ---------------------------------------------------------------------------
// Tiled bf16 WMMA GEMM:  out = epilogue(A(MxK) @ W(KxN) + bias)
// CONCAT: A is concat(h, x) f32 (K = 2*DIMD); else A is bf16, staged with
// GLOBAL_LOAD_ASYNC_TO_LDS_B32 (1 global dword -> 1 fragment dword).
// LDS staging writes directly in WMMA fragment order (ISA 7.12.2):
//   A 16x32 frag:  lane = r + 16*((k>>3)&1), dword = 4*((k>>4)&1) + (k&7)/2
//   B 32x16 frag:  lane = n + 16*((k>>3)&1), dword = 4*((k>>4)&1) + (k&7)/2
// K-chunk = 64 (two WMMA k-steps per barrier): 16 WMMAs/wave per stage.
// ---------------------------------------------------------------------------
template <int EPI, bool CONCAT>
__global__ __launch_bounds__(256)
void gemm_bf16_wmma(const unsigned short* __restrict__ A,
                    const float* __restrict__ hsrc,
                    const float* __restrict__ xsrc,
                    const float* __restrict__ W,
                    const float* __restrict__ bias,
                    unsigned short* __restrict__ outb,
                    float* __restrict__ outf,
                    int K, int N)
{
  __shared__ unsigned int sm[8192];          // 16KB A frags + 16KB B frags
  const int tid  = threadIdx.x;
  const int lane = tid & 31;
  const int w    = tid >> 5;
  const int wm   = w >> 2;                   // 0..1
  const int wn   = w & 3;                    // 0..3
  const int brow = blockIdx.y * 128;
  const int bcol = blockIdx.x * 128;

  const v8f vzero = {0.f, 0.f, 0.f, 0.f, 0.f, 0.f, 0.f, 0.f};
  v8f acc[4][2];
#pragma unroll
  for (int i = 0; i < 4; ++i)
#pragma unroll
    for (int j = 0; j < 2; ++j) acc[i][j] = vzero;

  for (int k0 = 0; k0 < K; k0 += 64) {
    __syncthreads();
    // ---- stage A tile (128 rows x 64 k) as bf16 fragment-order ----
#pragma unroll
    for (int j = 0; j < 16; ++j) {
      int p = tid + j * 256;                 // 0..4095 pair index
      int r = p >> 5;                        // 0..127
      int k = (p & 31) << 1;                 // even 0..62
      int grow = brow + r;
      int gk = k0 + k;
      int kc = k >> 5, kk = k & 31;
      int lt = (r & 15) + (((kk >> 3) & 1) << 4);
      int d  = (((kk >> 4) & 1) << 2) + ((kk & 7) >> 1);
      int dw = (((r >> 4) << 1) + kc) * 256 + (lt << 3) + d;
      if constexpr (CONCAT) {
        float f0, f1;
        if (gk < DIMD) {
          f0 = hsrc[(size_t)grow * DIMD + gk];
          f1 = hsrc[(size_t)grow * DIMD + gk + 1];
        } else {
          f0 = xsrc[(size_t)grow * DIMD + gk - DIMD];
          f1 = xsrc[(size_t)grow * DIMD + gk - DIMD + 1];
        }
        sm[dw] = pack2bf(f0, f1);
      } else {
        async_copy_b32((unsigned)(uintptr_t)&sm[dw],
                       A + (size_t)grow * K + gk);
      }
    }
    // ---- stage W tile (64 k x 128 n), f32 -> bf16, pairs along k ----
#pragma unroll
    for (int j = 0; j < 16; ++j) {
      int p = tid + j * 256;
      int n = p & 127;
      int k = (p >> 7) << 1;                 // even 0..62
      int gn = bcol + n;
      float w0 = W[(size_t)(k0 + k) * N + gn];
      float w1 = W[(size_t)(k0 + k + 1) * N + gn];
      if (k0 + 64 < K)                       // prefetch next chunk
        __builtin_prefetch(&W[(size_t)(k0 + 64 + k) * N + gn], 0, 1);
      int kc = k >> 5, kk = k & 31;
      int lt = (n & 15) + (((kk >> 3) & 1) << 4);
      int d  = (((kk >> 4) & 1) << 2) + ((kk & 7) >> 1);
      sm[4096 + (((n >> 4) << 1) + kc) * 256 + (lt << 3) + d] = pack2bf(w0, w1);
    }
    if constexpr (!CONCAT) wait_async0();
    __syncthreads();

    // ---- compute: 16 WMMAs per chunk per wave ----
#pragma unroll
    for (int kc = 0; kc < 2; ++kc) {
      FragB bfrag[2];
#pragma unroll
      for (int ns = 0; ns < 2; ++ns) {
        int base = 4096 + ((((wn * 2 + ns) << 1) + kc) << 8) + (lane << 3);
#pragma unroll
        for (int i = 0; i < 8; ++i) bfrag[ns].u[i] = sm[base + i];
      }
#pragma unroll
      for (int ms = 0; ms < 4; ++ms) {
        FragB afrag;
        int base = ((((wm * 4 + ms) << 1) + kc) << 8) + (lane << 3);
#pragma unroll
        for (int i = 0; i < 8; ++i) afrag.u[i] = sm[base + i];
#pragma unroll
        for (int ns = 0; ns < 2; ++ns) {
          acc[ms][ns] = __builtin_amdgcn_wmma_f32_16x16x32_bf16(
              false, afrag.v, false, bfrag[ns].v, (short)0, acc[ms][ns],
              false, false);
        }
      }
    }
  }

  // ---- epilogue (C layout: row = i + 8*(lane>=16), col = lane&15) ----
#pragma unroll
  for (int ms = 0; ms < 4; ++ms) {
#pragma unroll
    for (int ns = 0; ns < 2; ++ns) {
#pragma unroll
      for (int i = 0; i < 8; ++i) {
        int rowg = brow + wm * 64 + ms * 16 + i + ((lane >> 4) << 3);
        int colg = bcol + wn * 32 + ns * 16 + (lane & 15);
        float v = acc[ms][ns][i] + bias[colg];
        if constexpr (EPI == EPI_GELU) {
          v = 0.5f * v * (1.0f + erff(v * 0.70710678118f));
          outb[(size_t)rowg * N + colg] = f2bf(v);
        } else if constexpr (EPI == EPI_GATE) {
          float g  = 1.0f / (1.0f + __expf(-v));
          float hv = hsrc[(size_t)rowg * DIMD + colg];
          float xv = xsrc[(size_t)rowg * DIMD + colg];
          outb[(size_t)rowg * N + colg] = f2bf(g * hv + (1.0f - g) * xv);
        } else if constexpr (EPI == EPI_BF16) {
          outb[(size_t)rowg * N + colg] = f2bf(v);
        } else {
          outf[(size_t)rowg * N + colg] = v;
        }
      }
    }
  }
}

// ---------------------------------------------------------------------------
// Flash attention: grid (SEQ/128, BATCH*NHEADS), 256 threads (8 waves).
// Wave owns 16 q rows; workgroup shares 64-wide K/V tiles staged in LDS in
// B-fragment order (K tile staged via async global->LDS copies). Online
// softmax with shfl_xor(width=16) row reductions. P (C-layout f32) is
// re-laid-out via wave-private LDS into A-fragment bf16 for the PV WMMA.
// ---------------------------------------------------------------------------
__global__ __launch_bounds__(256)
void flash_attn_bf16(const unsigned short* __restrict__ q,
                     const unsigned short* __restrict__ kmat,
                     const unsigned short* __restrict__ vmat,
                     unsigned short* __restrict__ attn_out)
{
  __shared__ unsigned int sm[8192];          // 8KB Kfrag | 8KB Vfrag | 16KB P
  const int tid  = threadIdx.x;
  const int lane = tid & 31;
  const int w    = tid >> 5;
  const int b    = blockIdx.y >> 4;
  const int hd   = blockIdx.y & 15;
  const int qbase = blockIdx.x * 128 + w * 16;
  const size_t headoff = (size_t)hd * HDIM;

  const v8f vzero = {0.f, 0.f, 0.f, 0.f, 0.f, 0.f, 0.f, 0.f};

  // ---- Q strip (16 x 64) as two A-fragments, held in VGPRs ----
  FragB qf[2];
#pragma unroll
  for (int ks = 0; ks < 2; ++ks) {
#pragma unroll
    for (int d = 0; d < 8; ++d) {
      int r  = lane & 15;
      int kh = (lane >> 4) << 3;
      int k  = (d < 4) ? (kh + 2 * d) : (16 + kh + 2 * (d - 4));
      int dh = ks * 32 + k;
      qf[ks].u[d] = *(const unsigned int*)(
          q + (size_t)(b * SEQ + qbase + r) * DIMD + headoff + dh);
    }
  }

  v8f o[4];
  float mrow[8], lrow[8];
#pragma unroll
  for (int n = 0; n < 4; ++n) o[n] = vzero;
#pragma unroll
  for (int i = 0; i < 8; ++i) { mrow[i] = -1e30f; lrow[i] = 0.f; }

  for (int kv0 = 0; kv0 < SEQ; kv0 += 64) {
    __syncthreads();
    // ---- stage K tile (64 kv x 64 dh) as QK^T B-fragments, async ----
#pragma unroll
    for (int j = 0; j < 8; ++j) {
      int p   = tid + j * 256;
      int kvl = p >> 5;
      int dh  = (p & 31) << 1;
      int nsub = kvl >> 4, nn = kvl & 15;
      int ks = dh >> 5, kb = dh & 31;
      int lt = nn + (((kb >> 3) & 1) << 4);
      int d  = (((kb >> 4) & 1) << 2) + ((kb & 7) >> 1);
      int dw = ((nsub * 2 + ks) << 8) + (lt << 3) + d;
      async_copy_b32((unsigned)(uintptr_t)&sm[dw],
                     kmat + (size_t)(b * SEQ + kv0 + kvl) * DIMD + headoff + dh);
    }
    // ---- stage V tile (64 kv x 64 dh) as PV B-fragments (pair along kv) ----
#pragma unroll
    for (int j = 0; j < 8; ++j) {
      int p  = tid + j * 256;
      int kv = (p >> 6) << 1;
      int dh = p & 63;
      unsigned short v0 = vmat[(size_t)(b * SEQ + kv0 + kv) * DIMD + headoff + dh];
      unsigned short v1 = vmat[(size_t)(b * SEQ + kv0 + kv + 1) * DIMD + headoff + dh];
      int ks2 = kv >> 5, kp = kv & 31;
      int ns2 = dh >> 4, nn = dh & 15;
      int lt = nn + (((kp >> 3) & 1) << 4);
      int d  = (((kp >> 4) & 1) << 2) + ((kp & 7) >> 1);
      sm[2048 + ((ks2 * 4 + ns2) << 8) + (lt << 3) + d] =
          ((unsigned int)v1 << 16) | v0;
    }
    wait_async0();
    __syncthreads();

    // ---- scores S = (Q K^T) * 1/sqrt(Dh) ----
    v8f s[4];
#pragma unroll
    for (int n = 0; n < 4; ++n) {
      v8f t = vzero;
#pragma unroll
      for (int ks = 0; ks < 2; ++ks) {
        FragB kf;
        int base = ((n * 2 + ks) << 8) + (lane << 3);
#pragma unroll
        for (int i = 0; i < 8; ++i) kf.u[i] = sm[base + i];
        t = __builtin_amdgcn_wmma_f32_16x16x32_bf16(
            false, qf[ks].v, false, kf.v, (short)0, t, false, false);
      }
#pragma unroll
      for (int i = 0; i < 8; ++i) t[i] *= 0.125f;
      s[n] = t;
    }

    // ---- online softmax (rows per half-wave; reduce across 16 lanes) ----
#pragma unroll
    for (int i = 0; i < 8; ++i) {
      float t = fmaxf(fmaxf(s[0][i], s[1][i]), fmaxf(s[2][i], s[3][i]));
#pragma unroll
      for (int off = 1; off < 16; off <<= 1)
        t = fmaxf(t, __shfl_xor(t, off, 16));
      float mnew  = fmaxf(mrow[i], t);
      float alpha = __expf(mrow[i] - mnew);
      mrow[i] = mnew;
      lrow[i] *= alpha;
#pragma unroll
      for (int n = 0; n < 4; ++n) o[n][i] *= alpha;
      float rsum = 0.f;
#pragma unroll
      for (int n = 0; n < 4; ++n) {
        float pv = __expf(s[n][i] - mnew);
        s[n][i] = pv;
        rsum += pv;
      }
#pragma unroll
      for (int off = 1; off < 16; off <<= 1)
        rsum += __shfl_xor(rsum, off, 16);
      lrow[i] += rsum;
    }

    // ---- P: C-layout f32 -> A-fragment bf16 via wave-private LDS ----
    unsigned short* ps = (unsigned short*)sm;
    int pbase = 4096 + w * 512;
    int mh = (lane >> 4) << 3;
#pragma unroll
    for (int n = 0; n < 4; ++n) {
#pragma unroll
      for (int i = 0; i < 8; ++i) {
        int kcol = n * 16 + (lane & 15);
        int row  = mh + i;
        int ks = kcol >> 5, k = kcol & 31;
        int lt = row + (((k >> 3) & 1) << 4);
        int d  = (((k >> 4) & 1) << 2) + ((k & 7) >> 1);
        int dw = pbase + (ks << 8) + (lt << 3) + d;
        ps[dw * 2 + (k & 1)] = f2bf(s[n][i]);
      }
    }
    __syncthreads();

    // ---- O += P @ V ----
    FragB pf[2];
#pragma unroll
    for (int ks = 0; ks < 2; ++ks) {
      int base = pbase + (ks << 8) + (lane << 3);
#pragma unroll
      for (int i = 0; i < 8; ++i) pf[ks].u[i] = sm[base + i];
    }
#pragma unroll
    for (int n = 0; n < 4; ++n) {
#pragma unroll
      for (int ks = 0; ks < 2; ++ks) {
        FragB vf;
        int base = 2048 + ((ks * 4 + n) << 8) + (lane << 3);
#pragma unroll
        for (int i = 0; i < 8; ++i) vf.u[i] = sm[base + i];
        o[n] = __builtin_amdgcn_wmma_f32_16x16x32_bf16(
            false, pf[ks].v, false, vf.v, (short)0, o[n], false, false);
      }
    }
  }

  // ---- epilogue: O / l -> bf16 attn_out (B,S,H*Dh) ----
  int mo = (lane >> 4) << 3;
#pragma unroll
  for (int n = 0; n < 4; ++n) {
#pragma unroll
    for (int i = 0; i < 8; ++i) {
      int row = qbase + mo + i;
      int dh  = n * 16 + (lane & 15);
      attn_out[(size_t)(b * SEQ + row) * DIMD + headoff + dh] =
          f2bf(o[n][i] / lrow[i]);
    }
  }
}

// ---------------------------------------------------------------------------
extern "C" void kernel_launch(void* const* d_in, const int* in_sizes, int n_in,
                              void* d_out, int out_size, void* d_ws, size_t ws_size,
                              hipStream_t stream) {
  (void)in_sizes; (void)n_in; (void)out_size; (void)ws_size;
  const float* h   = (const float*)d_in[0];
  const float* x   = (const float*)d_in[1];
  const float* Wq  = (const float*)d_in[2];
  const float* bq  = (const float*)d_in[3];
  const float* Wk  = (const float*)d_in[4];
  const float* bk  = (const float*)d_in[5];
  const float* Wv  = (const float*)d_in[6];
  const float* bv  = (const float*)d_in[7];
  const float* Wo  = (const float*)d_in[8];
  const float* bo  = (const float*)d_in[9];
  const float* Wg1 = (const float*)d_in[10];
  const float* bg1 = (const float*)d_in[11];
  const float* Wg2 = (const float*)d_in[12];
  const float* bg2 = (const float*)d_in[13];

  char* ws = (char*)d_ws;
  unsigned short* hidden  = (unsigned short*)(ws);                       // 2 MB
  unsigned short* blended = (unsigned short*)(ws + (size_t)(2u  << 20)); // 8 MB
  unsigned short* qbuf    = (unsigned short*)(ws + (size_t)(10u << 20)); // 8 MB
  unsigned short* kbuf    = (unsigned short*)(ws + (size_t)(18u << 20)); // 8 MB
  unsigned short* vbuf    = (unsigned short*)(ws + (size_t)(26u << 20)); // 8 MB
  unsigned short* abuf    = (unsigned short*)(ws + (size_t)(34u << 20)); // 8 MB

  dim3 blk(256);
  // 1) hidden = GELU(concat(h,x) @ Wg1 + bg1)
  gemm_bf16_wmma<EPI_GELU, true><<<dim3(GHID / 128, MTOT / 128), blk, 0, stream>>>(
      nullptr, h, x, Wg1, bg1, hidden, nullptr, 2 * DIMD, GHID);
  // 2) gate = sigmoid(hidden @ Wg2 + bg2); blended = gate*h + (1-gate)*x
  gemm_bf16_wmma<EPI_GATE, false><<<dim3(DIMD / 128, MTOT / 128), blk, 0, stream>>>(
      hidden, h, x, Wg2, bg2, blended, nullptr, GHID, DIMD);
  // 3-5) Q, K, V projections
  gemm_bf16_wmma<EPI_BF16, false><<<dim3(DIMD / 128, MTOT / 128), blk, 0, stream>>>(
      blended, nullptr, nullptr, Wq, bq, qbuf, nullptr, DIMD, DIMD);
  gemm_bf16_wmma<EPI_BF16, false><<<dim3(DIMD / 128, MTOT / 128), blk, 0, stream>>>(
      blended, nullptr, nullptr, Wk, bk, kbuf, nullptr, DIMD, DIMD);
  gemm_bf16_wmma<EPI_BF16, false><<<dim3(DIMD / 128, MTOT / 128), blk, 0, stream>>>(
      blended, nullptr, nullptr, Wv, bv, vbuf, nullptr, DIMD, DIMD);
  // 6) flash attention
  flash_attn_bf16<<<dim3(SEQ / 128, BATCH * NHEADS), blk, 0, stream>>>(
      qbuf, kbuf, vbuf, abuf);
  // 7) out = attn_out @ Wo + bo  (f32)
  gemm_bf16_wmma<EPI_F32, false><<<dim3(DIMD / 128, MTOT / 128), blk, 0, stream>>>(
      abuf, nullptr, nullptr, Wo, bo, nullptr, (float*)d_out, DIMD, DIMD);
}
